// Glolbal_Attention_59373627899916
// MI455X (gfx1250) — matrix operand order repaired and checked by reading, not compiled, not run on hardware
//
#include <hip/hip_runtime.h>

typedef _Float16 half_t;
typedef __attribute__((ext_vector_type(4)))  float  v4f;
typedef __attribute__((ext_vector_type(8)))  float  v8f;
typedef __attribute__((ext_vector_type(4)))  half_t v4h;
typedef __attribute__((ext_vector_type(8)))  half_t v8h;
typedef __attribute__((ext_vector_type(16))) half_t v16h;

union AB { v16h v; v8h h[2]; };

#define NTOT   16384      // n = H*W
#define CH     128        // channels per head
#define BH     8          // B * heads
#define NB     64         // n-positions staged per stage
#define LDK    72         // padded LDS row stride (halves), 144B (16B aligned)
#define LDC    136        // padded LDS row stride for c=128 rows, 272B
#define NCHUNK 512        // n per workgroup -> 256 WGs total per kernel
#define NSPLIT (NTOT / NCHUNK)  // 32
#define SCALE  0.08838834764831845f  // 1/sqrt(128)

__device__ __forceinline__ float softplus(float x) {
    // numerically stable softplus
    return fmaxf(x, 0.0f) + log1pf(__expf(-fabsf(x)));
}

__device__ __forceinline__ void ld_frag16(AB& f, const half_t* p) {
    f.h[0] = *(const v8h*)(p);
    f.h[1] = *(const v8h*)(p + 8);
}

// ---------------------------------------------------------------------------
// Kernel 0: zero the kv / ksum accumulators in workspace (ws is poisoned).
// 8*128*128 + 8*128 = 132096 floats = 129 blocks * 256 threads * float4
// ---------------------------------------------------------------------------
__global__ __launch_bounds__(256) void zero_ws_kernel(float* __restrict__ ws) {
    int i = (blockIdx.x * 256 + threadIdx.x) * 4;
    v4f z = {};
    *(v4f*)(ws + i) = z;
}

// ---------------------------------------------------------------------------
// Kernel 1: kv[bh] += k_chunk @ v_chunk^T  (f16 WMMA, f32 accum, f32 atomics)
//           ksum[bh] += sum_n k_chunk
// grid = (NSPLIT, BH), block = 256 (8 waves). Wave w owns kv rows [16w,16w+16).
// ---------------------------------------------------------------------------
__global__ __launch_bounds__(256) void kv_kernel(const float* __restrict__ x,
                                                 float* __restrict__ ws_kv,
                                                 float* __restrict__ ws_ksum) {
    __shared__ half_t lds_k[CH * LDK];   // [c][n_local] f16, post-softplus
    __shared__ half_t lds_v[CH * LDK];   // [c][n_local] f16

    const int bh   = blockIdx.y;
    const int tid  = threadIdx.x;
    const int lane = tid & 31;
    const int wave = tid >> 5;       // 0..7
    const int l15  = lane & 15;
    const int lhi  = lane >> 4;      // 0 or 1
    const int n0   = blockIdx.x * NCHUNK;

    const float* kbase = x + (size_t)1 * BH * CH * NTOT + (size_t)bh * CH * NTOT;
    const float* vbase = x + (size_t)2 * BH * CH * NTOT + (size_t)bh * CH * NTOT;

    v8f acc[8];
    #pragma unroll
    for (int t = 0; t < 8; ++t) { v8f z = {}; acc[t] = z; }

    const int m_base = wave * 16;

    for (int s = 0; s < NCHUNK / NB; ++s) {
        const int ns = n0 + s * NB;
        __syncthreads();  // protect LDS reuse across stages
        // stage 128 rows x 64 cols of k and v (f32 -> f16), coalesced b128 loads
        #pragma unroll
        for (int it = 0; it < 8; ++it) {
            int i   = (tid + it * 256) * 4;  // element index in [row][col64]
            int row = i >> 6;
            int col = i & 63;
            v4f kf = *(const v4f*)(kbase + (size_t)row * NTOT + ns + col);
            v4f vf = *(const v4f*)(vbase + (size_t)row * NTOT + ns + col);
            // prefetch next stage's lines while this stage computes
            if (s + 1 < NCHUNK / NB) {
                __builtin_prefetch(kbase + (size_t)row * NTOT + ns + NB + col, 0, 1);
                __builtin_prefetch(vbase + (size_t)row * NTOT + ns + NB + col, 0, 1);
            }
            v4h kh, vh;
            kh.x = (half_t)softplus(kf.x); kh.y = (half_t)softplus(kf.y);
            kh.z = (half_t)softplus(kf.z); kh.w = (half_t)softplus(kf.w);
            vh.x = (half_t)vf.x; vh.y = (half_t)vf.y;
            vh.z = (half_t)vf.z; vh.w = (half_t)vf.w;
            *(v4h*)(&lds_k[row * LDK + col]) = kh;
            *(v4h*)(&lds_v[row * LDK + col]) = vh;
        }
        __syncthreads();

        // ksum partials (read-only on lds_k, safe alongside WMMA reads)
        if (tid < CH) {
            float srow = 0.0f;
            #pragma unroll 8
            for (int c2 = 0; c2 < NB; ++c2) srow += (float)lds_k[tid * LDK + c2];
            atomicAdd(&ws_ksum[bh * CH + tid], srow);
        }

        // 2 K-steps of 32; B fragments in an explicit even/odd double buffer
        // (both buffers live across each WMMA -> regalloc cannot merge them,
        //  loads for tile nt+2 fly while the WMMA for tile nt executes)
        #pragma unroll
        for (int kk = 0; kk < NB / 32; ++kk) {
            const int koff = kk * 32;
            // A fragment: k rows m_base+l15; contiguous K layout (k = 16*half + s), matches B
            const half_t* arow = &lds_k[(m_base + l15) * LDK + koff + lhi * 16];
            AB a;
            a.h[0] = *(const v8h*)(arow);
            a.h[1] = *(const v8h*)(arow + 8);
            const half_t* bbase = &lds_v[l15 * LDK + koff + lhi * 16];
            AB b0, b1;
            ld_frag16(b0, bbase);
            ld_frag16(b1, bbase + 16 * LDK);
            #pragma unroll
            for (int nt = 0; nt < 8; nt += 2) {
                acc[nt] = __builtin_amdgcn_wmma_f32_16x16x32_f16(
                    false, a.v, false, b0.v, (short)0, acc[nt], false, false);
                if (nt + 2 < 8) ld_frag16(b0, bbase + (size_t)(nt + 2) * 16 * LDK);
                acc[nt + 1] = __builtin_amdgcn_wmma_f32_16x16x32_f16(
                    false, a.v, false, b1.v, (short)0, acc[nt + 1], false, false);
                if (nt + 3 < 8) ld_frag16(b1, bbase + (size_t)(nt + 3) * 16 * LDK);
            }
        }
    }

    // accumulate partial kv into workspace (f32 atomics)
    float* kvb = ws_kv + (size_t)bh * CH * CH;
    #pragma unroll
    for (int nt = 0; nt < 8; ++nt) {
        #pragma unroll
        for (int r = 0; r < 8; ++r) {
            int m = m_base + r + lhi * 8;   // C/D layout: VGPR r -> M = r (+8 hi half)
            int d = nt * 16 + l15;          // N = lane & 15
            atomicAdd(&kvb[m * CH + d], acc[nt][r]);
        }
    }
}

// ---------------------------------------------------------------------------
// Kernel 2: out[d][n] = scale * z[n] * sum_c kv[c][d] * q[c][n]
// A = kv^T (staged transposed once, preloaded fragments), B = q^T (staged
// transposed per stage). Two n-tiles processed concurrently: independent
// WMMA accumulator chains + double-buffered B fragments.
// grid = (NSPLIT, BH), block = 256 (8 waves). Wave w owns d rows [16w,16w+16).
// ---------------------------------------------------------------------------
__global__ __launch_bounds__(256) void out_kernel(const float* __restrict__ x,
                                                  const float* __restrict__ ws_kv,
                                                  const float* __restrict__ ws_ksum,
                                                  float* __restrict__ out) {
    __shared__ half_t lds_kvT[CH * LDC];  // [d][c] f16
    __shared__ half_t lds_qT[NB * LDC];   // [n_local][c] f16, post-softplus
    __shared__ float  s_ksum[CH];
    __shared__ float  s_z[NB];

    const int bh   = blockIdx.y;
    const int tid  = threadIdx.x;
    const int lane = tid & 31;
    const int wave = tid >> 5;
    const int l15  = lane & 15;
    const int lhi  = lane >> 4;
    const int n0   = blockIdx.x * NCHUNK;

    const float* qbase = x + (size_t)bh * CH * NTOT;
    float*       obase = out + (size_t)bh * CH * NTOT;
    const float* kvb   = ws_kv + (size_t)bh * CH * CH;

    // stage kv transposed into LDS as f16 [d][c]
    #pragma unroll
    for (int it = 0; it < 16; ++it) {
        int e = (tid + it * 256) * 4;
        int c = e >> 7;
        int d = e & 127;
        v4f f = *(const v4f*)(kvb + e);
        lds_kvT[(d + 0) * LDC + c] = (half_t)f.x;
        lds_kvT[(d + 1) * LDC + c] = (half_t)f.y;
        lds_kvT[(d + 2) * LDC + c] = (half_t)f.z;
        lds_kvT[(d + 3) * LDC + c] = (half_t)f.w;
    }
    if (tid < CH) s_ksum[tid] = ws_ksum[bh * CH + tid];
    __syncthreads();

    // preload persistent A fragments (kv^T rows for this wave's d stripe)
    AB afrag[4];
    #pragma unroll
    for (int k0 = 0; k0 < 4; ++k0) {
        const half_t* ar = &lds_kvT[(wave * 16 + l15) * LDC + k0 * 32 + lhi * 16];
        afrag[k0].h[0] = *(const v8h*)(ar);
        afrag[k0].h[1] = *(const v8h*)(ar + 8);
    }

    for (int s = 0; s < NCHUNK / NB; ++s) {
        const int ns = n0 + s * NB;
        __syncthreads();  // protect lds_qT / s_z reuse
        // stage q transposed: read [c][n] coalesced f32, write [n][c] f16
        #pragma unroll
        for (int it = 0; it < 8; ++it) {
            int e  = (tid + it * 256) * 4;
            int c  = e >> 6;
            int nn = e & 63;
            v4f f = *(const v4f*)(qbase + (size_t)c * NTOT + ns + nn);
            if (s + 1 < NCHUNK / NB) {
                __builtin_prefetch(qbase + (size_t)c * NTOT + ns + NB + nn, 0, 1);
            }
            lds_qT[(nn + 0) * LDC + c] = (half_t)softplus(f.x);
            lds_qT[(nn + 1) * LDC + c] = (half_t)softplus(f.y);
            lds_qT[(nn + 2) * LDC + c] = (half_t)softplus(f.z);
            lds_qT[(nn + 3) * LDC + c] = (half_t)softplus(f.w);
        }
        __syncthreads();
        // normalizer z[n] = 1 / (scale * (q . ksum) + n)
        if (tid < NB) {
            float dot = 0.0f;
            #pragma unroll 8
            for (int c = 0; c < CH; ++c)
                dot += (float)lds_qT[tid * LDC + c] * s_ksum[c];
            s_z[tid] = 1.0f / (dot * SCALE + (float)NTOT);
        }
        __syncthreads();

        // two n-tiles in flight: independent accumulators, double-buffered B
        #pragma unroll
        for (int nt = 0; nt < NB / 16; nt += 2) {
            v8f acc0 = {}, acc1 = {};
            const half_t* bb0 = &lds_qT[((nt + 0) * 16 + l15) * LDC + lhi * 16];
            const half_t* bb1 = &lds_qT[((nt + 1) * 16 + l15) * LDC + lhi * 16];
            AB b0, b1;
            ld_frag16(b0, bb0);
            ld_frag16(b1, bb1);
            #pragma unroll
            for (int k0 = 0; k0 < 4; ++k0) {
                acc0 = __builtin_amdgcn_wmma_f32_16x16x32_f16(
                    false, afrag[k0].v, false, b0.v, (short)0, acc0, false, false);
                if (k0 < 3) ld_frag16(b0, bb0 + (k0 + 1) * 32);
                acc1 = __builtin_amdgcn_wmma_f32_16x16x32_f16(
                    false, afrag[k0].v, false, b1.v, (short)0, acc1, false, false);
                if (k0 < 3) ld_frag16(b1, bb1 + (k0 + 1) * 32);
            }
            float zv0 = s_z[(nt + 0) * 16 + l15] * SCALE;
            float zv1 = s_z[(nt + 1) * 16 + l15] * SCALE;
            #pragma unroll
            for (int r = 0; r < 8; ++r) {
                int d = wave * 16 + r + lhi * 8;
                obase[(size_t)d * NTOT + ns + (nt + 0) * 16 + l15] = acc0[r] * zv0;
                obase[(size_t)d * NTOT + ns + (nt + 1) * 16 + l15] = acc1[r] * zv1;
            }
        }
    }
}

// ---------------------------------------------------------------------------
extern "C" void kernel_launch(void* const* d_in, const int* in_sizes, int n_in,
                              void* d_out, int out_size, void* d_ws, size_t ws_size,
                              hipStream_t stream) {
    const float* x   = (const float*)d_in[0];
    float*       out = (float*)d_out;
    float*       ws  = (float*)d_ws;
    float* ws_kv   = ws;                      // 8*128*128 f32
    float* ws_ksum = ws + BH * CH * CH;       // 8*128 f32

    zero_ws_kernel<<<129, 256, 0, stream>>>(ws);

    dim3 grid(NSPLIT, BH);
    kv_kernel<<<grid, 256, 0, stream>>>(x, ws_kv, ws_ksum);
    out_kernel<<<grid, 256, 0, stream>>>(x, ws_kv, ws_ksum, out);
}
